// GraphTransformer_65335042507195
// MI455X (gfx1250) — matrix operand order repaired and checked by reading, not compiled, not run on hardware
//
#include <hip/hip_runtime.h>
#include <hip/hip_bf16.h>
#include <math.h>

// Problem constants (from reference)
#define NN   100000     // nodes
#define NE   256000     // edges
#define NG   4096       // graphs
#define EMB  300
#define NH   4
#define HD   75         // EMB / NH
#define EDIM 32
#define FEAT 256
#define NL   5
#define FFND 1200       // 4*EMB

typedef __attribute__((ext_vector_type(16))) _Float16 v16h;
typedef __attribute__((ext_vector_type(8)))  _Float16 v8h;
typedef __attribute__((ext_vector_type(8)))  float    v8f;

// ---------------------------------------------------------------------------
// WMMA GEMM:  C[M,N] = act( A[M,K] @ W[K,N] + bias (+ C if acc) )
// 256 threads = 8 waves. Block tile 128x64; each wave owns a 32x32 region
// (2x2 WMMA 16x16 tiles -> 4 v_wmma per K-chunk, fragments reused 2x).
// LDS tiles stored pre-swizzled in WMMA fragment order: each lane reads its
// 16-half fragment as one contiguous, 32B-aligned block (2x ds_load_b128).
// Double-buffered LDS: stage chunk c+1 while WMMAs consume chunk c
// (one barrier per chunk). Guarded path is branchless (clamp + cndmask).
// NOTE: all callers pass even K and even N (8B-aligned float2 loads OK).
// act: 0=none, 1=relu, 2=softplus
// ---------------------------------------------------------------------------
__global__ __launch_bounds__(256)
void wmma_gemm(const float* __restrict__ A, const float* __restrict__ W,
               const float* __restrict__ bias, float* __restrict__ C,
               int M, int N, int K, int act, int acc)
{
    // fragment-order tiles: [buf][16-row tile][lane][element]
    __shared__ __align__(32) _Float16 Af[2][8][32][16];   // 128 rows x 32 k
    __shared__ __align__(32) _Float16 Bf[2][4][32][16];   // 32 k x 64 cols

    const int tid  = threadIdx.x;
    const int wave = tid >> 5;
    const int lane = tid & 31;
    const int wm   = wave & 3;   // 0..3 : 32-row strip
    const int wn   = wave >> 2;  // 0..1 : 32-col strip
    const int l15  = lane & 15;
    const int hi   = lane >> 4;

    const int row0 = blockIdx.x * 128;
    const int col0 = blockIdx.y * 64;
    const bool fullMN = (row0 + 128 <= M) && (col0 + 64 <= N);

    v8f accs[2][2] = {};

    auto stage = [&](int k0, int buf) {
        if (fullMN && (k0 + 32 <= K)) {
            // A: 512 tasks (row m, 8-wide k-group g), 2 per thread.
            // k-group g -> (hi=g&1, e_base=(g>>1)*8): one 16B v8h store.
#pragma unroll
            for (int t = 0; t < 2; ++t) {
                int task = tid + t * 256;
                int m = task >> 2, g = task & 3;
                const float* src = A + (size_t)(row0 + m) * K + k0 + g * 8;
                v8h hv;
#pragma unroll
                for (int j = 0; j < 4; ++j) {
                    float2 f = *reinterpret_cast<const float2*>(src + 2 * j);
                    hv[2 * j]     = (_Float16)f.x;
                    hv[2 * j + 1] = (_Float16)f.y;
                }
                *(v8h*)&Af[buf][m >> 4][(g & 1) * 16 + (m & 15)][(g >> 1) * 8] = hv;
            }
            // B: task = (k row, 8-wide n-group), 1 per thread
            {
                int k = tid >> 3, ng = tid & 7;
                const float* src = W + (size_t)(k0 + k) * N + col0 + ng * 8;
#pragma unroll
                for (int j = 0; j < 4; ++j) {
                    float2 f = *reinterpret_cast<const float2*>(src + 2 * j);
                    int n0 = ng * 8 + 2 * j;
                    Bf[buf][n0 >> 4][((k >> 3) & 1) * 16 + (n0 & 15)][((k >> 4) << 3) + (k & 7)] = (_Float16)f.x;
                    int n1 = n0 + 1;
                    Bf[buf][n1 >> 4][((k >> 3) & 1) * 16 + (n1 & 15)][((k >> 4) << 3) + (k & 7)] = (_Float16)f.y;
                }
            }
            // keep a prefetch flowing two chunks ahead (global_prefetch_b8)
            if (k0 + 64 < K) {
                int pr = row0 + (tid >> 1);
                __builtin_prefetch(&A[(size_t)pr * K + k0 + 64], 0, 1);
            }
        } else {
            // branchless guarded path: clamp indices, unconditional load,
            // cndmask to zero when out of range
#pragma unroll
            for (int t = 0; t < 2; ++t) {
                int task = tid + t * 256;
                int m = task >> 2, g = task & 3;
                int gr  = row0 + m;
                int grc = gr < M ? gr : M - 1;
                const float* srcr = A + (size_t)grc * K;
#pragma unroll
                for (int j = 0; j < 8; ++j) {
                    int kk  = g * 8 + j;
                    int gk  = k0 + kk;
                    int gkc = gk < K ? gk : K - 1;
                    float v = srcr[gkc];
                    v = (gr < M && gk < K) ? v : 0.0f;
                    int h2 = (kk >> 3) & 1;
                    int e  = ((kk >> 4) << 3) + (kk & 7);
                    Af[buf][m >> 4][h2 * 16 + (m & 15)][e] = (_Float16)v;
                }
            }
            {
                int k = tid >> 3, ng = tid & 7;
                int gk  = k0 + k;
                int gkc = gk < K ? gk : K - 1;
                const float* srcr = W + (size_t)gkc * N;
#pragma unroll
                for (int j = 0; j < 8; ++j) {
                    int n = ng * 8 + j, gc = col0 + n;
                    int gcc = gc < N ? gc : N - 1;
                    float v = srcr[gcc];
                    v = (gk < K && gc < N) ? v : 0.0f;
                    Bf[buf][n >> 4][((k >> 3) & 1) * 16 + (n & 15)][((k >> 4) << 3) + (k & 7)] = (_Float16)v;
                }
            }
        }
    };

    const int nchunks = (K + 31) / 32;
    stage(0, 0);
    for (int c = 0; c < nchunks; ++c) {
        __syncthreads();                         // stage(c) visible
        if (c + 1 < nchunks) stage((c + 1) * 32, (c + 1) & 1);  // overlap

        const int buf = c & 1;
        // contiguous 32B fragment reads (2x ds_load_b128 each)
        v16h a0 = *(const v16h*)&Af[buf][wm * 2 + 0][lane][0];
        v16h a1 = *(const v16h*)&Af[buf][wm * 2 + 1][lane][0];
        v16h b0 = *(const v16h*)&Bf[buf][wn * 2 + 0][lane][0];
        v16h b1 = *(const v16h*)&Bf[buf][wn * 2 + 1][lane][0];

        accs[0][0] = __builtin_amdgcn_wmma_f32_16x16x32_f16(
            false, a0, false, b0, (short)0, accs[0][0], false, false);
        accs[0][1] = __builtin_amdgcn_wmma_f32_16x16x32_f16(
            false, a0, false, b1, (short)0, accs[0][1], false, false);
        accs[1][0] = __builtin_amdgcn_wmma_f32_16x16x32_f16(
            false, a1, false, b0, (short)0, accs[1][0], false, false);
        accs[1][1] = __builtin_amdgcn_wmma_f32_16x16x32_f16(
            false, a1, false, b1, (short)0, accs[1][1], false, false);
    }

    // epilogue: C/D layout -> VGPR r holds M = r + 8*hi, N = lane&15
#pragma unroll
    for (int sm = 0; sm < 2; ++sm)
#pragma unroll
        for (int sn = 0; sn < 2; ++sn) {
#pragma unroll
            for (int r = 0; r < 8; ++r) {
                int row = row0 + wm * 32 + sm * 16 + hi * 8 + r;
                int col = col0 + wn * 32 + sn * 16 + l15;
                if (row < M && col < N) {
                    float v = accs[sm][sn][r];
                    if (bias) v += bias[col];
                    if (acc)  v += C[(size_t)row * N + col];
                    if (act == 1)      v = fmaxf(v, 0.0f);
                    else if (act == 2) v = (v > 20.0f) ? v : log1pf(__expf(v));
                    C[(size_t)row * N + col] = v;
                }
            }
        }
}

// ---------------------------------------------------------------------------
// element-wise helpers
// ---------------------------------------------------------------------------
__global__ void fill_kernel(float* __restrict__ p, float v, long n) {
    long i = (long)blockIdx.x * blockDim.x + threadIdx.x;
    if (i < n) p[i] = v;
}

__global__ void node_embed(const int* __restrict__ x, const float* __restrict__ xe1,
                           const float* __restrict__ xe2, float* __restrict__ h) {
    long i = (long)blockIdx.x * blockDim.x + threadIdx.x;
    if (i >= (long)NN * EMB) return;
    int n = (int)(i / EMB), c = (int)(i % EMB);
    h[i] = xe1[(size_t)x[n * 2 + 0] * EMB + c] + xe2[(size_t)x[n * 2 + 1] * EMB + c];
}

__global__ void edge_embed(const int* __restrict__ ea, const float* __restrict__ ee1,
                           const float* __restrict__ ee2, float* __restrict__ etmp) {
    long i = (long)blockIdx.x * blockDim.x + threadIdx.x;
    if (i >= (long)NE * EDIM) return;
    int e = (int)(i / EDIM), c = (int)(i % EDIM);
    etmp[i] = ee1[(size_t)ea[e * 2 + 0] * EDIM + c] + ee2[(size_t)ea[e * 2 + 1] * EDIM + c];
}

// ---------------------------------------------------------------------------
// attention: segment softmax over incoming edges per (node, head)
// ---------------------------------------------------------------------------
__device__ __forceinline__ void atomicMaxF(float* addr, float val) {
    int* ia = (int*)addr;
    int old = __float_as_int(*addr);
    while (__int_as_float(old) < val) {
        int assumed = old;
        old = atomicCAS(ia, assumed, __float_as_int(val));
        if (old == assumed) break;
    }
}

__global__ void attn_alpha(const float* __restrict__ q, const float* __restrict__ k,
                           const float* __restrict__ ep, const int* __restrict__ src,
                           const int* __restrict__ dst, float* __restrict__ alpha) {
    long i = (long)blockIdx.x * blockDim.x + threadIdx.x;
    if (i >= (long)NE * NH) return;
    int e = (int)(i / NH), hh = (int)(i % NH);
    const float* qd = q  + (size_t)dst[e] * EMB + hh * HD;
    const float* ks = k  + (size_t)src[e] * EMB + hh * HD;
    const float* ee = ep + (size_t)e      * EMB + hh * HD;
    float s = 0.0f;
    for (int d = 0; d < HD; ++d) s += qd[d] * (ks[d] + ee[d]);
    alpha[i] = s * 0.11547005383792516f;   // 1/sqrt(75)
}

__global__ void attn_amax(const float* __restrict__ alpha, const int* __restrict__ dst,
                          float* __restrict__ amax) {
    long i = (long)blockIdx.x * blockDim.x + threadIdx.x;
    if (i >= (long)NE * NH) return;
    int e = (int)(i / NH), hh = (int)(i % NH);
    atomicMaxF(&amax[(size_t)dst[e] * NH + hh], alpha[i]);
}

__global__ void attn_exp(const float* __restrict__ alpha, const float* __restrict__ amax,
                         const int* __restrict__ dst, float* __restrict__ ex,
                         float* __restrict__ den) {
    long i = (long)blockIdx.x * blockDim.x + threadIdx.x;
    if (i >= (long)NE * NH) return;
    int e = (int)(i / NH), hh = (int)(i % NH);
    float v = __expf(alpha[i] - amax[(size_t)dst[e] * NH + hh]);
    ex[i] = v;
    atomicAdd(&den[(size_t)dst[e] * NH + hh], v);
}

__global__ void attn_scatter(const float* __restrict__ vmat, const float* __restrict__ ep,
                             const float* __restrict__ ex, const float* __restrict__ den,
                             const int* __restrict__ src, const int* __restrict__ dst,
                             float* __restrict__ out) {
    long i = (long)blockIdx.x * blockDim.x + threadIdx.x;
    if (i >= (long)NE * EMB) return;
    int e = (int)(i / EMB), c = (int)(i % EMB);
    int hh = c / HD;
    int d  = dst[e];
    float a = ex[(size_t)e * NH + hh] / (den[(size_t)d * NH + hh] + 1e-16f);
    float m = (vmat[(size_t)src[e] * EMB + c] + ep[(size_t)e * EMB + c]) * a;
    atomicAdd(&out[(size_t)d * EMB + c], m);
}

// ---------------------------------------------------------------------------
// residual + layernorm: out = LN(x + r) * g + b ; one wave32 per row
// ---------------------------------------------------------------------------
__global__ __launch_bounds__(32)
void resid_ln(const float* __restrict__ x, const float* __restrict__ r,
              const float* __restrict__ g, const float* __restrict__ b,
              float* __restrict__ out, int M, int C) {
    int row = blockIdx.x;
    if (row >= M) return;
    int lane = threadIdx.x;
    float vals[10];
    int   cnt = 0;
    float s = 0.0f;
    for (int c = lane; c < C; c += 32) {
        float v = x[(size_t)row * C + c] + r[(size_t)row * C + c];
        vals[cnt++] = v;
        s += v;
    }
    for (int off = 16; off; off >>= 1) s += __shfl_xor(s, off, 32);
    float mean = s / (float)C;
    float vs = 0.0f;
    for (int i = 0; i < cnt; ++i) { float d = vals[i] - mean; vs += d * d; }
    for (int off = 16; off; off >>= 1) vs += __shfl_xor(vs, off, 32);
    float rstd = rsqrtf(vs / (float)C + 1e-5f);
    cnt = 0;
    for (int c = lane; c < C; c += 32)
        out[(size_t)row * C + c] = (vals[cnt++] - mean) * rstd * g[c] + b[c];
}

// ---------------------------------------------------------------------------
// global mean pooling
// ---------------------------------------------------------------------------
__global__ void pool_scatter(const float* __restrict__ h, const int* __restrict__ batch,
                             float* __restrict__ hgm) {
    long i = (long)blockIdx.x * blockDim.x + threadIdx.x;
    if (i >= (long)NN * EMB) return;
    int n = (int)(i / EMB), c = (int)(i % EMB);
    atomicAdd(&hgm[(size_t)batch[n] * EMB + c], h[i]);
}
__global__ void pool_count(const int* __restrict__ batch, float* __restrict__ cnt) {
    long i = (long)blockIdx.x * blockDim.x + threadIdx.x;
    if (i >= NN) return;
    atomicAdd(&cnt[batch[i]], 1.0f);
}
__global__ void pool_div(float* __restrict__ hgm, const float* __restrict__ cnt) {
    long i = (long)blockIdx.x * blockDim.x + threadIdx.x;
    if (i >= (long)NG * EMB) return;
    hgm[i] /= fmaxf(cnt[i / EMB], 1.0f);
}

// ---------------------------------------------------------------------------
// host orchestration
// ---------------------------------------------------------------------------
static inline dim3 gemm_grid(int M, int N) { return dim3((M + 127) / 128, (N + 63) / 64); }
static inline int  blks(long n, int t)     { return (int)((n + t - 1) / t); }

extern "C" void kernel_launch(void* const* d_in, const int* in_sizes, int n_in,
                              void* d_out, int out_size, void* d_ws, size_t ws_size,
                              hipStream_t stream) {
    const int*   x     = (const int*)d_in[0];
    const int*   ei    = (const int*)d_in[1];   // [2,E]: src=ei, dst=ei+NE
    const int*   ea    = (const int*)d_in[2];
    const int*   batch = (const int*)d_in[3];
    const float* xe1 = (const float*)d_in[4],  *xe2 = (const float*)d_in[5];
    const float* ee1 = (const float*)d_in[6],  *ee2 = (const float*)d_in[7];
    const float* epw = (const float*)d_in[8],  *epb = (const float*)d_in[9];
    const float* wq  = (const float*)d_in[10], *bq  = (const float*)d_in[11];
    const float* wk  = (const float*)d_in[12], *bk  = (const float*)d_in[13];
    const float* wv  = (const float*)d_in[14], *bv  = (const float*)d_in[15];
    const float* we  = (const float*)d_in[16];
    const float* wsk = (const float*)d_in[17], *bsk = (const float*)d_in[18];
    const float* g1  = (const float*)d_in[19], *b1  = (const float*)d_in[20];
    const float* g2  = (const float*)d_in[21], *b2  = (const float*)d_in[22];
    const float* fw1 = (const float*)d_in[23], *fb1 = (const float*)d_in[24];
    const float* fw2 = (const float*)d_in[25], *fb2 = (const float*)d_in[26];
    const float* ftw = (const float*)d_in[27], *ftb = (const float*)d_in[28];
    const float* pw1 = (const float*)d_in[29], *pb1 = (const float*)d_in[30];
    const float* pw2 = (const float*)d_in[31], *pb2 = (const float*)d_in[32];

    const int* src = ei;
    const int* dst = ei + NE;

    size_t off = 0;
    auto alloc = [&](size_t floats) -> float* {
        float* p = (float*)((char*)d_ws + off);
        off += ((floats * sizeof(float) + 255) / 256) * 256;
        return p;
    };
    float* h     = alloc((size_t)NN * EMB);
    float* q     = alloc((size_t)NN * EMB);
    float* k     = alloc((size_t)NN * EMB);
    float* v     = alloc((size_t)NN * EMB);
    float* outb  = alloc((size_t)NN * EMB);
    float* ffnh  = alloc((size_t)NN * FFND);
    float* eproj = alloc((size_t)NE * EMB);
    float* ef    = alloc((size_t)NE * EDIM);
    float* etmp  = alloc((size_t)NE * EDIM);
    float* alpha = alloc((size_t)NE * NH);
    float* ex    = alloc((size_t)NE * NH);
    float* amax  = alloc((size_t)NN * NH);
    float* den   = alloc((size_t)NN * NH);
    float* hgm   = alloc((size_t)NG * EMB);
    float* cnt   = alloc((size_t)NG);
    float* ptmp  = alloc((size_t)NG * 128);
    (void)ws_size;

    float* hg_feat = (float*)d_out;                 // [NG, FEAT]
    float* pred    = hg_feat + (size_t)NG * FEAT;   // [NG, 2]

    node_embed<<<blks((long)NN * EMB, 256), 256, 0, stream>>>(x, xe1, xe2, h);
    edge_embed<<<blks((long)NE * EDIM, 256), 256, 0, stream>>>(ea, ee1, ee2, etmp);
    wmma_gemm<<<gemm_grid(NE, EDIM), 256, 0, stream>>>(etmp, epw, epb, ef,
                                                       NE, EDIM, EDIM, 0, 0);

    for (int l = 0; l < NL; ++l) {
        const float* wql = wq + (size_t)l * EMB * EMB;
        const float* wkl = wk + (size_t)l * EMB * EMB;
        const float* wvl = wv + (size_t)l * EMB * EMB;
        const float* wel = we + (size_t)l * EDIM * EMB;
        const float* wsl = wsk + (size_t)l * EMB * EMB;

        wmma_gemm<<<gemm_grid(NN, EMB), 256, 0, stream>>>(h, wql, bq + l * EMB, q,
                                                          NN, EMB, EMB, 0, 0);
        wmma_gemm<<<gemm_grid(NN, EMB), 256, 0, stream>>>(h, wkl, bk + l * EMB, k,
                                                          NN, EMB, EMB, 0, 0);
        wmma_gemm<<<gemm_grid(NN, EMB), 256, 0, stream>>>(h, wvl, bv + l * EMB, v,
                                                          NN, EMB, EMB, 0, 0);
        wmma_gemm<<<gemm_grid(NE, EMB), 256, 0, stream>>>(ef, wel, nullptr, eproj,
                                                          NE, EMB, EDIM, 0, 0);

        fill_kernel<<<blks((long)NN * NH, 256), 256, 0, stream>>>(amax, -1e30f, (long)NN * NH);
        fill_kernel<<<blks((long)NN * NH, 256), 256, 0, stream>>>(den, 0.0f, (long)NN * NH);
        fill_kernel<<<blks((long)NN * EMB, 256), 256, 0, stream>>>(outb, 0.0f, (long)NN * EMB);
        attn_alpha<<<blks((long)NE * NH, 256), 256, 0, stream>>>(q, k, eproj, src, dst, alpha);
        attn_amax<<<blks((long)NE * NH, 256), 256, 0, stream>>>(alpha, dst, amax);
        attn_exp<<<blks((long)NE * NH, 256), 256, 0, stream>>>(alpha, amax, dst, ex, den);
        attn_scatter<<<blks((long)NE * EMB, 256), 256, 0, stream>>>(v, eproj, ex, den,
                                                                    src, dst, outb);

        wmma_gemm<<<gemm_grid(NN, EMB), 256, 0, stream>>>(h, wsl, bsk + l * EMB, outb,
                                                          NN, EMB, EMB, 0, 1);
        resid_ln<<<NN, 32, 0, stream>>>(h, outb, g1 + l * EMB, b1 + l * EMB, h, NN, EMB);

        wmma_gemm<<<gemm_grid(NN, FFND), 256, 0, stream>>>(
            h, fw1 + (size_t)l * EMB * FFND, fb1 + l * FFND, ffnh, NN, FFND, EMB, 1, 0);
        wmma_gemm<<<gemm_grid(NN, EMB), 256, 0, stream>>>(
            ffnh, fw2 + (size_t)l * FFND * EMB, fb2 + l * EMB, q, NN, EMB, FFND, 0, 0);
        resid_ln<<<NN, 32, 0, stream>>>(h, q, g2 + l * EMB, b2 + l * EMB, h, NN, EMB);
    }

    fill_kernel<<<blks((long)NG * EMB, 256), 256, 0, stream>>>(hgm, 0.0f, (long)NG * EMB);
    fill_kernel<<<blks(NG, 256), 256, 0, stream>>>(cnt, 0.0f, NG);
    pool_scatter<<<blks((long)NN * EMB, 256), 256, 0, stream>>>(h, batch, hgm);
    pool_count<<<blks(NN, 256), 256, 0, stream>>>(batch, cnt);
    pool_div<<<blks((long)NG * EMB, 256), 256, 0, stream>>>(hgm, cnt);

    wmma_gemm<<<gemm_grid(NG, FEAT), 256, 0, stream>>>(hgm, ftw, ftb, hg_feat,
                                                       NG, FEAT, EMB, 0, 0);
    wmma_gemm<<<gemm_grid(NG, 128), 256, 0, stream>>>(hg_feat, pw1, pb1, ptmp,
                                                      NG, 128, FEAT, 2, 0);  // softplus
    wmma_gemm<<<gemm_grid(NG, 2), 256, 0, stream>>>(ptmp, pw2, pb2, pred,
                                                    NG, 2, 128, 0, 0);
}
